// MultiheadAttention_23914377904490
// MI455X (gfx1250) — matrix operand order repaired and checked by reading, not compiled, not run on hardware
//
#include <hip/hip_runtime.h>
#include <hip/hip_bf16.h>

// ---------------------------------------------------------------------------
// Borzoi relative-position multihead attention for MI455X (gfx1250, wave32).
// H=8 heads, K=64, V=192, T=2048, C=1536, F=32 position features.
// All GEMMs via v_wmma_f32_16x16x32_bf16. Rel-position logits computed via
// suffix-sum tables (central-mask features are step functions of |d|).
// Attention K/V tiles staged into LDS via global_load_async_to_lds_b128.
// ---------------------------------------------------------------------------

typedef __bf16 bf16;
typedef __attribute__((ext_vector_type(8)))  bf16  v8bf;
typedef __attribute__((ext_vector_type(16))) bf16  v16bf;
typedef __attribute__((ext_vector_type(8)))  float v8f;
typedef int v4i __attribute__((vector_size(16)));   // matches builtin prototype

#define HEADS 8
#define KDIM  64
#define VDIM  192
#define TSEQ  2048
#define CDIM  1536
#define QSCALE 0.125f   // 64^-0.5

#ifndef __has_builtin
#define __has_builtin(x) 0
#endif
#if __has_builtin(__builtin_amdgcn_global_load_async_to_lds_b128)
#define HAVE_ASYNC_LDS 1
#define LDS_AS __attribute__((address_space(3)))
#define GLB_AS __attribute__((address_space(1)))
#else
#define HAVE_ASYNC_LDS 0
#endif

__device__ __forceinline__ unsigned short f2bf(float f) {
  unsigned u = __float_as_uint(f);
  u += 0x7FFFu + ((u >> 16) & 1u);           // round-to-nearest-even
  return (unsigned short)(u >> 16);
}

__device__ __forceinline__ v16bf cat16(v8bf lo, v8bf hi) {
  v16bf r;
#pragma unroll
  for (int i = 0; i < 8; ++i) { r[i] = lo[i]; r[i + 8] = hi[i]; }
  return r;
}

// A-fragment (16x32 bf16, MxK): lane = row M; lanes<16 hold K={0..7,16..23},
// lanes>=16 hold K={8..15,24..31}. Two 16B loads per lane.
__device__ __forceinline__ v16bf load_a_frag(const bf16* base, int ld, int lane) {
  const int half = lane >> 4, m = lane & 15;
  const bf16* p = base + (size_t)m * ld + half * 8;
  v8bf lo = *(const v8bf*)(const void*)p;
  v8bf hi = *(const v8bf*)(const void*)(p + 16);
  return cat16(lo, hi);
}

// B-fragment (32x16 bf16, KxN): lane = col N; lanes<16 hold K={0..7,16..23},
// lanes>=16 hold K={8..15,24..31}.  baseT is row-major [N, Kd].
__device__ __forceinline__ v16bf load_b_frag(const bf16* baseT, int ld, int lane) {
  const int half = lane >> 4, n = lane & 15;
  const bf16* p = baseT + (size_t)n * ld + half * 8;
  v8bf lo = *(const v8bf*)(const void*)p;
  v8bf hi = *(const v8bf*)(const void*)(p + 16);
  return cat16(lo, hi);
}

__device__ __forceinline__ v8f wmma_bf16(v16bf a, v16bf b, v8f c) {
  return __builtin_amdgcn_wmma_f32_16x16x32_bf16(false, a, false, b,
                                                 (short)0, c, false, false);
}

// 16-byte global->LDS copy: async-to-LDS when the builtin exists, otherwise
// a plain load/store pair (global_load_b128 + ds_store_b128).
__device__ __forceinline__ void cp16(unsigned short* dst_lds,
                                     const unsigned short* src_glob) {
#if HAVE_ASYNC_LDS
  __builtin_amdgcn_global_load_async_to_lds_b128(
      (GLB_AS v4i*)src_glob, (LDS_AS v4i*)dst_lds, 0, 0);
#else
  *(v8bf*)(void*)dst_lds = *(const v8bf*)(const void*)src_glob;
#endif
}

__device__ __forceinline__ void async_fence() {
#if HAVE_ASYNC_LDS
  asm volatile("s_wait_asynccnt 0x0" ::: "memory");
#endif
}

// ---------------------------------------------------------------------------
// Elementwise converts / packs
// ---------------------------------------------------------------------------
__global__ void k_f32_to_bf16(const float* __restrict__ in,
                              unsigned short* __restrict__ out, int n) {
  int i = blockIdx.x * 256 + threadIdx.x;
  if (i < n) out[i] = f2bf(in[i]);
}

// W [Kd=1536, N] fp32 -> WT [N, 1536] bf16
__global__ void k_transpose_w(const float* __restrict__ W,
                              unsigned short* __restrict__ WT, int N) {
  size_t i = (size_t)blockIdx.x * 256 + threadIdx.x;   // over N*1536
  int k = (int)(i % CDIM);
  int n = (int)(i / CDIM);
  WT[i] = f2bf(W[(size_t)k * N + n]);
}

// qf/kf [T, 512] fp32 -> qw/kb [H, T, 64] bf16 (q scaled + r_w_bias folded)
__global__ void k_pack_qk(const float* __restrict__ qf, const float* __restrict__ kf,
                          const float* __restrict__ rwb,
                          unsigned short* __restrict__ qw,
                          unsigned short* __restrict__ kb) {
  size_t i = (size_t)blockIdx.x * 256 + threadIdx.x;   // H*T*64 = 1048576
  int k = (int)(i & 63);
  size_t ht = i >> 6;
  int t = (int)(ht & (TSEQ - 1));
  int h = (int)(ht >> 11);
  size_t src = (size_t)t * (HEADS * KDIM) + h * KDIM + k;
  qw[i] = f2bf(qf[src] * QSCALE + rwb[h * KDIM + k]);
  kb[i] = f2bf(kf[src]);
}

// vf [T, 1536] fp32 -> vT [H, 192, T] bf16 (transposed for B-fragment loads)
__global__ void k_pack_v(const float* __restrict__ vf,
                         unsigned short* __restrict__ vT) {
  size_t i = (size_t)blockIdx.x * 256 + threadIdx.x;   // H*192*T = 3145728
  int t = (int)(i & (TSEQ - 1));
  size_t hv = i >> 11;
  int vc = (int)(hv % VDIM);
  int h  = (int)(hv / VDIM);
  vT[i] = f2bf(vf[(size_t)t * CDIM + h * VDIM + vc]);
}

// Threshold table: t(|d|) = #{nb : cw[nb] <= |d|}, cw[nb] = pr^(nb+1) - 1
__global__ void k_ttable(unsigned char* __restrict__ tt) {
  int ad = blockIdx.x * 256 + threadIdx.x;   // 0..2047
  float pr = __expf(__logf((float)(TSEQ + 1)) / 16.0f);
  float cw = pr;
  int t = 0;
#pragma unroll
  for (int nb = 0; nb < 16; ++nb) {
    if (cw - 1.0f <= (float)ad) ++t;
    cw *= pr;
  }
  tt[ad] = (unsigned char)t;
}

// Per (h,t): u[f] = sum_k (qf*s + r_r_bias)[k] * Wr[f, h*64+k], then suffix
// sums: S[h][t][0..16] = S1 (f=m..15 of u[0..15]), S[h][t][17..33] = S2.
__global__ void k_compute_S(const float* __restrict__ qf,
                            const float* __restrict__ Wr,
                            const float* __restrict__ rrb,
                            float* __restrict__ S) {
  int h = blockIdx.y;
  int t = blockIdx.x * 256 + threadIdx.x;
  __shared__ float w[32][KDIM];
  for (int i = threadIdx.x; i < 32 * KDIM; i += 256)
    w[i >> 6][i & 63] = Wr[(size_t)(i >> 6) * (HEADS * KDIM) + h * KDIM + (i & 63)];
  __syncthreads();

  float u[32];
#pragma unroll
  for (int f = 0; f < 32; ++f) u[f] = 0.0f;
  for (int k = 0; k < KDIM; ++k) {
    float qv = qf[(size_t)t * (HEADS * KDIM) + h * KDIM + k] * QSCALE
             + rrb[h * KDIM + k];
#pragma unroll
    for (int f = 0; f < 32; ++f) u[f] += qv * w[f][k];
  }
  float* out = S + ((size_t)h * TSEQ + t) * 34;
  float s1 = 0.0f, s2 = 0.0f;
  out[16] = 0.0f; out[33] = 0.0f;
#pragma unroll
  for (int m = 15; m >= 0; --m) {
    s1 += u[m];      out[m]      = s1;
    s2 += u[16 + m]; out[17 + m] = s2;
  }
}

// ---------------------------------------------------------------------------
// bf16 WMMA GEMM: C[M,N] = A[M,Kd] @ BT[N,Kd]^T (+ bias)
// One wave per block computes a 32x64 tile (2 row-frags x 4 col-frags,
// 8 WMMA per 32-wide k-step).  grid = (M/32, N/64), block = 32.
// ---------------------------------------------------------------------------
__global__ void k_gemm_bf16(const unsigned short* __restrict__ A,
                            const unsigned short* __restrict__ BT,
                            const float* __restrict__ bias,
                            float* __restrict__ C, int N, int Kd) {
  const int lane = threadIdx.x & 31;
  const int half = lane >> 4, nl = lane & 15;
  const int row0 = blockIdx.x * 32;
  const int n0   = blockIdx.y * 64;
  const bf16* Ab = (const bf16*)A + (size_t)row0 * Kd;
  const bf16* Bb = (const bf16*)BT + (size_t)n0 * Kd;

  v8f acc[2][4];
#pragma unroll
  for (int mt = 0; mt < 2; ++mt)
#pragma unroll
    for (int nt = 0; nt < 4; ++nt) acc[mt][nt] = v8f{};

  for (int kd = 0; kd < Kd; kd += 32) {
    __builtin_prefetch(Ab + kd + 64, 0, 0);               // global_prefetch_b8
    __builtin_prefetch(Bb + kd + 64, 0, 0);
    v16bf a0 = load_a_frag(Ab + kd,                   Kd, lane);
    v16bf a1 = load_a_frag(Ab + (size_t)16 * Kd + kd, Kd, lane);
#pragma unroll
    for (int nt = 0; nt < 4; ++nt) {
      v16bf b = load_b_frag(Bb + (size_t)(nt * 16) * Kd + kd, Kd, lane);
      acc[0][nt] = wmma_bf16(a0, b, acc[0][nt]);
      acc[1][nt] = wmma_bf16(a1, b, acc[1][nt]);
    }
  }

#pragma unroll
  for (int nt = 0; nt < 4; ++nt) {
    int n = n0 + nt * 16 + nl;
    float bv = bias ? bias[n] : 0.0f;
#pragma unroll
    for (int mt = 0; mt < 2; ++mt)
#pragma unroll
      for (int r = 0; r < 8; ++r)
        C[(size_t)(row0 + mt * 16 + half * 8 + r) * N + n] = acc[mt][nt][r] + bv;
  }
}

// ---------------------------------------------------------------------------
// Flash attention with rel-position suffix-sum tables.
// grid = (T/64, H); block = 128 (4 waves); wave w handles 16 query rows.
// K (32x64) and V^T (192x32) tiles staged cooperatively into LDS per block.
// ---------------------------------------------------------------------------
__global__ void k_attn(const unsigned short* __restrict__ qw,
                       const unsigned short* __restrict__ kbf,
                       const unsigned short* __restrict__ vT,
                       const float* __restrict__ S,
                       const unsigned char* __restrict__ tt,
                       unsigned short* __restrict__ ao) {
  const int h    = blockIdx.y;
  const int tid  = threadIdx.x;
  const int w    = tid >> 5;
  const int lane = tid & 31;
  const int half = lane >> 4, nl = lane & 15;
  const int q0   = blockIdx.x * 64 + w * 16;

  __shared__ unsigned char ltt[TSEQ];
  __shared__ float lS[4][16][34];
  __shared__ __align__(16) unsigned short lP[4][16 * 32];
  __shared__ __align__(16) unsigned short lK[32 * KDIM];   // 4 KB
  __shared__ __align__(16) unsigned short lV[VDIM * 32];   // 12 KB

  for (int i = tid; i < TSEQ; i += 128) ltt[i] = tt[i];
  {
    const float* Sp = S + ((size_t)h * TSEQ + q0) * 34;
    for (int i = lane; i < 16 * 34; i += 32) lS[w][i / 34][i % 34] = Sp[i];
  }
  __syncthreads();

  const bf16* Aq = (const bf16*)qw + ((size_t)h * TSEQ + q0) * KDIM;
  v16bf aq0 = load_a_frag(Aq,      KDIM, lane);
  v16bf aq1 = load_a_frag(Aq + 32, KDIM, lane);

  const unsigned short* gK = kbf + (size_t)h * TSEQ * KDIM;
  const unsigned short* gV = vT + (size_t)h * VDIM * TSEQ;

  v8f o[12];
#pragma unroll
  for (int vt = 0; vt < 12; ++vt) o[vt] = v8f{};
  float m8[8], l8[8];
#pragma unroll
  for (int r = 0; r < 8; ++r) { m8[r] = -1e30f; l8[r] = 0.0f; }

  for (int j0 = 0; j0 < TSEQ; j0 += 32) {
    // ---- cooperative staging of K / V tiles into LDS ----
    __syncthreads();                       // previous iteration's reads done
    {
      const unsigned short* src = gK + (size_t)j0 * KDIM;  // contiguous 4 KB
      for (int i = tid; i < 256; i += 128) cp16(&lK[i * 8], src + i * 8);
      for (int i = tid; i < 768; i += 128) {               // 192 rows x 64 B
        int vc = i >> 2, seg = i & 3;
        cp16(&lV[vc * 32 + seg * 8], gV + (size_t)vc * TSEQ + j0 + seg * 8);
      }
    }
    async_fence();
    __syncthreads();

    // ---- logits: 16q x 32j, two 16-col WMMA tiles, K=64 = 2 k-steps ----
    v8f s0 = {}, s1 = {};
    {
      const bf16* Kt = (const bf16*)lK;
      v16bf b;
      b = load_b_frag(Kt,                 KDIM, lane); s0 = wmma_bf16(aq0, b, s0);
      b = load_b_frag(Kt + 32,            KDIM, lane); s0 = wmma_bf16(aq1, b, s0);
      b = load_b_frag(Kt + 16 * KDIM,     KDIM, lane); s1 = wmma_bf16(aq0, b, s1);
      b = load_b_frag(Kt + 16 * KDIM + 32, KDIM, lane); s1 = wmma_bf16(aq1, b, s1);
    }

    // ---- rel-position add (suffix-sum lookups) + row max ----
    float rm[8];
#pragma unroll
    for (int r = 0; r < 8; ++r) {
      const int mrow = half * 8 + r;
      const int q = q0 + mrow;
      const float* Sr = lS[w][mrow];
      int j = j0 + nl;
      int d = j - q; int ad = d < 0 ? -d : d; int t = ltt[ad];
      float sg = (d > 0) ? 1.0f : ((d < 0) ? -1.0f : 0.0f);
      s0[r] += Sr[t] + sg * Sr[17 + t];
      j += 16;
      d = j - q; ad = d < 0 ? -d : d; t = ltt[ad];
      sg = (d > 0) ? 1.0f : ((d < 0) ? -1.0f : 0.0f);
      s1[r] += Sr[t] + sg * Sr[17 + t];
      float v = fmaxf(s0[r], s1[r]);
      v = fmaxf(v, __shfl_xor(v, 1));
      v = fmaxf(v, __shfl_xor(v, 2));
      v = fmaxf(v, __shfl_xor(v, 4));
      v = fmaxf(v, __shfl_xor(v, 8));
      rm[r] = v;
    }

    // ---- online softmax update ----
#pragma unroll
    for (int r = 0; r < 8; ++r) {
      float mn = fmaxf(m8[r], rm[r]);
      float al = __expf(m8[r] - mn);
      m8[r] = mn;
      float p0 = __expf(s0[r] - mn);
      float p1 = __expf(s1[r] - mn);
      s0[r] = p0; s1[r] = p1;
      float ps = p0 + p1;
      ps += __shfl_xor(ps, 1);
      ps += __shfl_xor(ps, 2);
      ps += __shfl_xor(ps, 4);
      ps += __shfl_xor(ps, 8);
      l8[r] = l8[r] * al + ps;
#pragma unroll
      for (int vt = 0; vt < 12; ++vt) o[vt][r] *= al;
    }

    // ---- transpose P (C-layout -> A-layout) via per-wave LDS ----
#pragma unroll
    for (int r = 0; r < 8; ++r) {
      const int mrow = half * 8 + r;
      lP[w][mrow * 32 + nl]      = f2bf(s0[r]);
      lP[w][mrow * 32 + 16 + nl] = f2bf(s1[r]);
    }
    asm volatile("s_wait_dscnt 0" ::: "memory");   // wave-local LDS RAW
    v16bf pa = load_a_frag((const bf16*)lP[w], 32, lane);

    // ---- O += P @ V : 12 tiles across V=192, B-frags from LDS ----
#pragma unroll
    for (int vt = 0; vt < 12; ++vt) {
      v16bf bv = load_b_frag((const bf16*)lV + (size_t)(vt * 16) * 32, 32, lane);
      o[vt] = wmma_bf16(pa, bv, o[vt]);
    }
  }

  // ---- normalize and emit ao[T, H*V] bf16 ----
  float inv[8];
#pragma unroll
  for (int r = 0; r < 8; ++r) inv[r] = 1.0f / l8[r];
#pragma unroll
  for (int vt = 0; vt < 12; ++vt)
#pragma unroll
    for (int r = 0; r < 8; ++r) {
      int q = q0 + half * 8 + r;
      ao[(size_t)q * CDIM + h * VDIM + vt * 16 + nl] = f2bf(o[vt][r] * inv[r]);
    }
}

// ---------------------------------------------------------------------------
// Host launcher
// ---------------------------------------------------------------------------
extern "C" void kernel_launch(void* const* d_in, const int* in_sizes, int n_in,
                              void* d_out, int out_size, void* d_ws, size_t ws_size,
                              hipStream_t stream) {
  (void)in_sizes; (void)n_in; (void)out_size; (void)ws_size;
  const float* x   = (const float*)d_in[0];
  const float* Wq  = (const float*)d_in[1];
  const float* Wk  = (const float*)d_in[2];
  const float* Wv  = (const float*)d_in[3];
  const float* Wr  = (const float*)d_in[4];
  const float* rwb = (const float*)d_in[5];
  const float* rrb = (const float*)d_in[6];
  const float* Wo  = (const float*)d_in[7];
  const float* bo  = (const float*)d_in[8];
  float* out = (float*)d_out;

  char* ws = (char*)d_ws;
  size_t off = 0;
  auto alloc = [&](size_t bytes) {
    size_t r = off; off += (bytes + 255) & ~(size_t)255; return r;
  };
  const size_t o_xbf = alloc((size_t)TSEQ * CDIM * 2);
  const size_t o_wqT = alloc((size_t)512 * CDIM * 2);
  const size_t o_wkT = alloc((size_t)512 * CDIM * 2);
  const size_t o_wvT = alloc((size_t)CDIM * CDIM * 2);
  const size_t o_woT = alloc((size_t)CDIM * CDIM * 2);
  const size_t o_qf  = alloc((size_t)TSEQ * 512 * 4);
  const size_t o_kf  = alloc((size_t)TSEQ * 512 * 4);
  const size_t o_vf  = alloc((size_t)TSEQ * CDIM * 4);
  const size_t o_qw  = alloc((size_t)HEADS * TSEQ * KDIM * 2);
  const size_t o_kb  = alloc((size_t)HEADS * TSEQ * KDIM * 2);
  const size_t o_vT  = alloc((size_t)HEADS * VDIM * TSEQ * 2);
  const size_t o_S   = alloc((size_t)HEADS * TSEQ * 34 * 4);
  const size_t o_tt  = alloc(2048);
  const size_t o_ao  = alloc((size_t)TSEQ * CDIM * 2);

  unsigned short* xbf = (unsigned short*)(ws + o_xbf);
  unsigned short* wqT = (unsigned short*)(ws + o_wqT);
  unsigned short* wkT = (unsigned short*)(ws + o_wkT);
  unsigned short* wvT = (unsigned short*)(ws + o_wvT);
  unsigned short* woT = (unsigned short*)(ws + o_woT);
  float* qf = (float*)(ws + o_qf);
  float* kf = (float*)(ws + o_kf);
  float* vf = (float*)(ws + o_vf);
  unsigned short* qw = (unsigned short*)(ws + o_qw);
  unsigned short* kb = (unsigned short*)(ws + o_kb);
  unsigned short* vT = (unsigned short*)(ws + o_vT);
  float* Sbuf = (float*)(ws + o_S);
  unsigned char* tt = (unsigned char*)(ws + o_tt);
  unsigned short* ao = (unsigned short*)(ws + o_ao);

  // Stage 0: converts / transposes
  k_f32_to_bf16<<<(TSEQ * CDIM + 255) / 256, 256, 0, stream>>>(x, xbf, TSEQ * CDIM);
  k_transpose_w<<<(512 * CDIM) / 256, 256, 0, stream>>>(Wq, wqT, 512);
  k_transpose_w<<<(512 * CDIM) / 256, 256, 0, stream>>>(Wk, wkT, 512);
  k_transpose_w<<<(CDIM * CDIM) / 256, 256, 0, stream>>>(Wv, wvT, CDIM);
  k_transpose_w<<<(CDIM * CDIM) / 256, 256, 0, stream>>>(Wo, woT, CDIM);

  // Stage 1: projections (WMMA)
  k_gemm_bf16<<<dim3(TSEQ / 32, 512 / 64), 32, 0, stream>>>(xbf, wqT, nullptr, qf, 512, CDIM);
  k_gemm_bf16<<<dim3(TSEQ / 32, 512 / 64), 32, 0, stream>>>(xbf, wkT, nullptr, kf, 512, CDIM);
  k_gemm_bf16<<<dim3(TSEQ / 32, CDIM / 64), 32, 0, stream>>>(xbf, wvT, nullptr, vf, CDIM, CDIM);

  // Stage 2: packing + rel-position tables
  k_pack_qk<<<(HEADS * TSEQ * KDIM) / 256, 256, 0, stream>>>(qf, kf, rwb, qw, kb);
  k_compute_S<<<dim3(TSEQ / 256, HEADS), 256, 0, stream>>>(qf, Wr, rrb, Sbuf);
  k_pack_v<<<(HEADS * VDIM * TSEQ) / 256, 256, 0, stream>>>(vf, vT);
  k_ttable<<<TSEQ / 256, 256, 0, stream>>>(tt);

  // Stage 3: flash attention (WMMA)
  k_attn<<<dim3(TSEQ / 64, HEADS), 128, 0, stream>>>(qw, kb, vT, Sbuf, tt, ao);

  // Stage 4: output projection (WMMA) + bias
  k_gemm_bf16<<<dim3(TSEQ / 32, CDIM / 64), 32, 0, stream>>>(ao, woT, bo, out, CDIM, CDIM);
}
